// DepthwiseSeparableConv_11184094839306
// MI455X (gfx1250) — compile-verified, matrix operand order checked by reading, not run on hardware
//
#include <hip/hip_runtime.h>
#include <hip/hip_bf16.h>

#define Bn   16
#define Cn   128
#define Hn   128
#define Wn   128
#define OUTn 256

typedef __attribute__((ext_vector_type(8)))  __bf16 v8bf;
typedef __attribute__((ext_vector_type(16))) __bf16 v16bf;
typedef __attribute__((ext_vector_type(8)))  float  v8f;

// round-to-nearest-even float -> bf16 bits
__device__ __forceinline__ unsigned short f2bf_rn(float f) {
    unsigned int u = __float_as_uint(f);
    unsigned int r = u + 0x7FFFu + ((u >> 16) & 1u);
    return (unsigned short)(r >> 16);
}
__device__ __forceinline__ float bf2f(unsigned short h) {
    return __uint_as_float(((unsigned int)h) << 16);
}

// Load a 16x32 (A: MxK) or 32x16 (B: KxN, stored pixel-major [n][k]) bf16
// fragment from LDS. Per-lane: 8 contiguous bf16 at k-base, 8 more at +16,
// matching the CDNA5 wave32 WMMA bf16 VGPR layout.
__device__ __forceinline__ v16bf load_frag(const unsigned short* base, int row0, int lane, int kc) {
    const int m = row0 + (lane & 15);
    const int k = kc * 32 + ((lane >> 4) << 3);
    const v8bf a = *(const v8bf*)(base + m * Cn + k);
    const v8bf b = *(const v8bf*)(base + m * Cn + k + 16);
    return __builtin_shufflevector(a, b, 0,1,2,3,4,5,6,7,8,9,10,11,12,13,14,15);
}

__global__ void __launch_bounds__(256)
dwsep_pw_fused_kernel(const float* __restrict__ x,
                      const float* __restrict__ colk,   // (C,1,3,1)
                      const float* __restrict__ rowk,   // (C,1,1,3)
                      const float* __restrict__ pw,     // (OUT, C)
                      float* __restrict__ out)          // (B, OUT, H, W)
{
    extern __shared__ unsigned short smem[];
    unsigned short* w_hi = smem;                     // OUTn*Cn
    unsigned short* w_lo = w_hi + OUTn * Cn;         // OUTn*Cn
    unsigned short* y_hi = w_lo + OUTn * Cn;         // 256 px * Cn
    unsigned short* y_lo = y_hi + 256 * Cn;          // 256 px * Cn

    const int tid = threadIdx.x;
    const int b   = blockIdx.y;
    const int h0  = blockIdx.x * 2;   // 2 output rows per workgroup

    // ---- stage pointwise weights into LDS as bf16 hi/lo split ----
    {
        int i = tid * 128;
        const int e = i + 128;
        for (; i < e; ++i) {
            const float v = pw[i];
            const unsigned short h = f2bf_rn(v);
            w_hi[i] = h;
            w_lo[i] = f2bf_rn(v - bf2f(h));
        }
    }

    // ---- depthwise separable 3x3 (fp32), write y tile as bf16 hi/lo ----
    {
        const int c     = tid >> 1;
        const int wbase = (tid & 1) * 64;
        const float rk0 = rowk[c * 3 + 0], rk1 = rowk[c * 3 + 1], rk2 = rowk[c * 3 + 2];
        const float ck0 = colk[c * 3 + 0], ck1 = colk[c * 3 + 1], ck2 = colk[c * 3 + 2];
        const float* xp = x + ((size_t)b * Cn + c) * (size_t)(Hn * Wn);
        for (int w = wbase; w < wbase + 64; ++w) {
            float t[4];
            #pragma unroll
            for (int rr = 0; rr < 4; ++rr) {
                const int row = h0 - 1 + rr;
                float a = 0.0f;
                if (row >= 0 && row < Hn) {
                    const float* rp = xp + row * Wn;
                    if (w > 0)      a += rk0 * rp[w - 1];
                    a += rk1 * rp[w];
                    if (w < Wn - 1) a += rk2 * rp[w + 1];
                }
                t[rr] = a;
            }
            const float y0 = ck0 * t[0] + ck1 * t[1] + ck2 * t[2];
            const float y1 = ck0 * t[1] + ck1 * t[2] + ck2 * t[3];
            const unsigned short hb0 = f2bf_rn(y0);
            y_hi[w * Cn + c]         = hb0;
            y_lo[w * Cn + c]         = f2bf_rn(y0 - bf2f(hb0));
            const unsigned short hb1 = f2bf_rn(y1);
            y_hi[(128 + w) * Cn + c] = hb1;
            y_lo[(128 + w) * Cn + c] = f2bf_rn(y1 - bf2f(hb1));
        }
    }

    __syncthreads();

    // ---- GEMM: out[256,256px] = W[256,128] @ y[128,256px], bf16x3 WMMA ----
    const int wv   = tid >> 5;
    const int lane = tid & 31;
    for (int mt = wv; mt < 16; mt += 8) {
        v16bf Ahi[4], Alo[4];
        #pragma unroll
        for (int kc = 0; kc < 4; ++kc) {
            Ahi[kc] = load_frag(w_hi, mt * 16, lane, kc);
            Alo[kc] = load_frag(w_lo, mt * 16, lane, kc);
        }
        for (int nt = 0; nt < 16; ++nt) {
            v8f acc = {};
            #pragma unroll
            for (int kc = 0; kc < 4; ++kc) {
                const v16bf Bhi = load_frag(y_hi, nt * 16, lane, kc);
                const v16bf Blo = load_frag(y_lo, nt * 16, lane, kc);
                acc = __builtin_amdgcn_wmma_f32_16x16x32_bf16(
                          false, Ahi[kc], false, Bhi, (short)0, acc, false, false);
                acc = __builtin_amdgcn_wmma_f32_16x16x32_bf16(
                          false, Ahi[kc], false, Blo, (short)0, acc, false, false);
                acc = __builtin_amdgcn_wmma_f32_16x16x32_bf16(
                          false, Alo[kc], false, Bhi, (short)0, acc, false, false);
            }
            // D layout: lane(0..15)=N, VGPR r -> M = r (+8 for lanes 16..31)
            const int n     = nt * 16 + (lane & 15);
            const int rloc  = n >> 7;          // row within 2-row tile
            const int wcol  = n & 127;
            const int mbase = mt * 16 + ((lane >> 4) << 3);
            float* op = out + ((((size_t)b * OUTn + mbase) * Hn + (h0 + rloc)) * Wn + wcol);
            #pragma unroll
            for (int r = 0; r < 8; ++r)
                op[(size_t)r * (Hn * Wn)] = acc[r];
        }
    }
}

extern "C" void kernel_launch(void* const* d_in, const int* in_sizes, int n_in,
                              void* d_out, int out_size, void* d_ws, size_t ws_size,
                              hipStream_t stream) {
    const float* x    = (const float*)d_in[0];
    const float* colk = (const float*)d_in[1];
    const float* rowk = (const float*)d_in[2];
    const float* pw   = (const float*)d_in[3];
    float* out        = (float*)d_out;

    const size_t shmem = (size_t)(2 * OUTn * Cn + 2 * 256 * Cn) * sizeof(unsigned short); // 256 KiB
    (void)hipFuncSetAttribute(reinterpret_cast<const void*>(dwsep_pw_fused_kernel),
                              hipFuncAttributeMaxDynamicSharedMemorySize, (int)shmem);

    dim3 grid(Hn / 2, Bn);   // 64 row-blocks x 16 batches
    dim3 block(256);
    dwsep_pw_fused_kernel<<<grid, block, shmem, stream>>>(x, colk, rowk, pw, out);
}
